// DeepHedging_NN_d_42210938585851
// MI455X (gfx1250) — matrix operand order repaired and checked by reading, not compiled, not run on hardware
//
#include <hip/hip_runtime.h>

typedef __attribute__((ext_vector_type(16))) _Float16 v16h;
typedef __attribute__((ext_vector_type(8)))  _Float16 v8h;
typedef __attribute__((ext_vector_type(8)))  float    v8f;

#define NSTEP 250
#define INDIM 4

// ---------------------------------------------------------------------------
// Prep: build folded f16 weights in workspace.
//   WhhH    : f16 [192][64]  = W_hh (recurrent weights, row-major)
//   WxPackH : f16 [192][32]  = [ W_x(4) | w_d | bias_i(+b_hh for r,z) | 0... ]
//     W_x  = W_ih[:, :64] @ W_in                      (192x4)
//     w_d  = W_ih[:, 64]
//     bias = W_ih[:, :64] @ b_in + b_ih (+ b_hh for gates < 128)
// ---------------------------------------------------------------------------
__global__ void gru_prep_kernel(const float* __restrict__ W_in,
                                const float* __restrict__ b_in,
                                const float* __restrict__ W_ih,
                                const float* __restrict__ W_hh,
                                const float* __restrict__ b_ih,
                                const float* __restrict__ b_hh,
                                _Float16* __restrict__ WhhH,
                                _Float16* __restrict__ WxPackH) {
  int g = threadIdx.x;
  if (g >= 192) return;
  for (int k = 0; k < 64; ++k)
    WhhH[g * 64 + k] = (_Float16)W_hh[g * 64 + k];

  float wx0 = 0.f, wx1 = 0.f, wx2 = 0.f, wx3 = 0.f;
  float bi = b_ih[g];
  for (int j = 0; j < 64; ++j) {
    float w = W_ih[g * 65 + j];
    bi += w * b_in[j];
    wx0 += w * W_in[j * 4 + 0];
    wx1 += w * W_in[j * 4 + 1];
    wx2 += w * W_in[j * 4 + 2];
    wx3 += w * W_in[j * 4 + 3];
  }
  if (g < 128) bi += b_hh[g];  // fuse recurrent bias for r,z gates
  float wd = W_ih[g * 65 + 64];

  _Float16* row = WxPackH + g * 32;
  for (int c = 0; c < 32; ++c) row[c] = (_Float16)0.0f;
  row[0] = (_Float16)wx0;
  row[1] = (_Float16)wx1;
  row[2] = (_Float16)wx2;
  row[3] = (_Float16)wx3;
  row[4] = (_Float16)wd;
  row[5] = (_Float16)bi;
}

// ---------------------------------------------------------------------------
// Main persistent-cell kernel
// ---------------------------------------------------------------------------
__device__ __forceinline__ v8f wmma16(v16h a, v16h b, v8f c) {
  return __builtin_amdgcn_wmma_f32_16x16x32_f16(false, a, false, b, (short)0, c,
                                                false, false);
}

#if __has_builtin(__builtin_amdgcn_tanhf)
__device__ __forceinline__ float tanh_f(float x) {
  return __builtin_amdgcn_tanhf(x);  // v_tanh_f32 (CDNA5 trans op)
}
#else
__device__ __forceinline__ float tanh_f(float x) {
  float e = __expf(2.0f * x);
  return 1.0f - 2.0f * __builtin_amdgcn_rcpf(e + 1.0f);
}
#endif

__device__ __forceinline__ float sigmoid_f(float x) {
  // sigmoid(x) = 0.5 + 0.5 * tanh(x/2): 1 trans op instead of exp+rcp
  return fmaf(0.5f, tanh_f(0.5f * x), 0.5f);
}

// A-tile load (16x32 f16, A layout): lane holds row M = lane%16,
// elems 0..7 -> K = 8*(lane/16)+0..7, elems 8..15 -> +16.
__device__ __forceinline__ v16h load_atile(const _Float16* p) {
  v8h a0 = *(const v8h*)(p);
  v8h a1 = *(const v8h*)(p + 16);
  v16h r;
#pragma unroll
  for (int e = 0; e < 8; ++e) {
    r[e] = a0[e];
    r[e + 8] = a1[e];
  }
  return r;
}

__global__ __launch_bounds__(256) void gru_persist_kernel(
    const float* __restrict__ X, const float* __restrict__ b_hh,
    const float* __restrict__ W_out, const float* __restrict__ b_out,
    const _Float16* __restrict__ WhhH, const _Float16* __restrict__ WxPackH,
    float* __restrict__ out) {
  const int lane = threadIdx.x & 31;
  const int wave = threadIdx.x >> 5;
  const int lo = lane & 15;
  const int hi = lane >> 4;  // 0 for lanes 0-15, 1 for lanes 16-31
  const int simBase = (blockIdx.x * 8 + wave) * 16;

  // ---- constant weight tiles, register resident -------------------------
  v16h Ah[12][2];  // h-side: tile t covers gates 16t..16t+15; kt: hid 0-31/32-63
#pragma unroll
  for (int t = 0; t < 12; ++t)
#pragma unroll
    for (int kt = 0; kt < 2; ++kt)
      Ah[t][kt] = load_atile(WhhH + (16 * t + lo) * 64 + kt * 32 + hi * 8);

  v16h Ax[12];  // packed i-side (W_x | w_d | bias)
#pragma unroll
  for (int t = 0; t < 12; ++t)
    Ax[t] = load_atile(WxPackH + (16 * t + lo) * 32 + hi * 8);

  v8f biasHN[4];  // b_hh for the n-block, in D layout
#pragma unroll
  for (int j = 0; j < 4; ++j)
    biasHN[j] = *(const v8f*)(b_hh + 128 + 16 * j + 8 * hi);

  v8f woutD[4];  // W_out in D layout for the output projection
#pragma unroll
  for (int j = 0; j < 4; ++j)
    woutD[j] = *(const v8f*)(W_out + 16 * j + 8 * hi);
  const float bo = b_out[0];

  // ---- recurrent state ---------------------------------------------------
  // hD: fp32, D layout: lane holds h[sim = lane%16][hid = 16j + 8*hi + v]
  float hD[4][8];
#pragma unroll
  for (int j = 0; j < 4; ++j)
#pragma unroll
    for (int v = 0; v < 8; ++v) hD[j][v] = 0.0f;

  // hB: f16, B layout: lane holds h[sim = lane%16][hid = 32*kt + 8*hi + e (+8 for e >= 8)]
  v16h hB[2];
#pragma unroll
  for (int e = 0; e < 16; ++e) {
    hB[0][e] = (_Float16)0.0f;
    hB[1][e] = (_Float16)0.0f;
  }
  float d = 0.0f;

  const float* xp = X + (size_t)(simBase + lo) * NSTEP * INDIM;
  float* op = out + (size_t)(simBase + lo) * NSTEP;
  const bool lohalf = (hi == 0);

  // i-side B operand: rows [x0..x3, d, 1, 0...]; elems 5..15 are constant.
  v16h xdB;
#pragma unroll
  for (int e = 0; e < 16; ++e) xdB[e] = (_Float16)0.0f;
  xdB[5] = (_Float16)(lohalf ? 1.0f : 0.0f);

  // Software pipeline: xv holds x for the current step.
  float4 xv = *(const float4*)(xp);

  for (int t = 0; t < NSTEP; ++t) {
    // Issue next step's X load early; its latency hides behind this step.
    int tn = (t + 1 == NSTEP) ? 0 : t + 1;  // branch-free clamp
    float4 xnext = *(const float4*)(xp + tn * 4);
    // WGP-scope prefetch one cacheline (8 steps) ahead: fills near caches,
    // consumed by this same wave shortly.
    __builtin_prefetch(xp + (t + 8) * 4, 0, 3);

    // ---- update the dynamic lanes of the i-side B operand ---------------
    xdB[0] = (_Float16)(lohalf ? xv.x : 0.0f);
    xdB[1] = (_Float16)(lohalf ? xv.y : 0.0f);
    xdB[2] = (_Float16)(lohalf ? xv.z : 0.0f);
    xdB[3] = (_Float16)(lohalf ? xv.w : 0.0f);
    xdB[4] = (_Float16)(lohalf ? d : 0.0f);

    const v8f zero8 = {};

    // ---- gate tiles: j indexes 16 gates of each of r,z,n ----------------
#pragma unroll
    for (int j = 0; j < 4; ++j) {
      v8f accR = wmma16(Ax[j], xdB, zero8);           // x + d + bias (i+h fused)
      accR = wmma16(Ah[j][0], hB[0], accR);           // + h @ W_hh_r
      accR = wmma16(Ah[j][1], hB[1], accR);

      v8f accZ = wmma16(Ax[4 + j], xdB, zero8);
      accZ = wmma16(Ah[4 + j][0], hB[0], accZ);
      accZ = wmma16(Ah[4 + j][1], hB[1], accZ);

      v8f accIN = wmma16(Ax[8 + j], xdB, zero8);      // i_n (x + d + bias_i)
      v8f accHN = wmma16(Ah[8 + j][0], hB[0], biasHN[j]);  // h_n (+ b_hh)
      accHN = wmma16(Ah[8 + j][1], hB[1], accHN);

#pragma unroll
      for (int v = 0; v < 8; ++v) {
        float r = sigmoid_f(accR[v]);
        float z = sigmoid_f(accZ[v]);
        float n = tanh_f(fmaf(r, accHN[v], accIN[v]));
        hD[j][v] = fmaf(z, hD[j][v] - n, n);  // (1-z)*n + z*h
      }
    }

    // ---- output projection d = h_new @ W_out.T + b_out ------------------
    float acc = 0.0f;
#pragma unroll
    for (int j = 0; j < 4; ++j)
#pragma unroll
      for (int v = 0; v < 8; ++v) acc = fmaf(woutD[j][v], hD[j][v], acc);
    acc += __shfl_xor(acc, 16, 32);  // combine the two half-wave partials
    d = acc + bo;
    if (lane < 16) op[t] = d;

    // ---- rebuild hB (f16, B layout) from this lane's own hD (shuffles unused) ------
#pragma unroll
    for (int kt = 0; kt < 2; ++kt) {
#pragma unroll
      for (int v = 0; v < 8; ++v) {
        float a = hD[2 * kt][v];
        float b = hD[2 * kt + 1][v];
        float axc = __shfl_xor(a, 16, 32);
        float bxc = __shfl_xor(b, 16, 32);
        float elo = a;  // elems 0..7  -> K = 8*hi + v
        float ehi = b;  // elems 8..15 -> K = 16 + 8*hi + v
        hB[kt][v] = (_Float16)elo;
        hB[kt][v + 8] = (_Float16)ehi;
      }
    }

    xv = xnext;
  }
}

// ---------------------------------------------------------------------------
extern "C" void kernel_launch(void* const* d_in, const int* in_sizes, int n_in,
                              void* d_out, int out_size, void* d_ws,
                              size_t ws_size, hipStream_t stream) {
  const float* X = (const float*)d_in[0];
  const float* W_in = (const float*)d_in[1];
  const float* b_in = (const float*)d_in[2];
  const float* W_ih = (const float*)d_in[3];
  const float* W_hh = (const float*)d_in[4];
  const float* b_ih = (const float*)d_in[5];
  const float* b_hh = (const float*)d_in[6];
  const float* W_out = (const float*)d_in[7];
  const float* b_out = (const float*)d_in[8];
  float* out = (float*)d_out;

  _Float16* WhhH = (_Float16*)d_ws;                                  // 24576 B
  _Float16* WxPackH = (_Float16*)((char*)d_ws + 192 * 64 * 2);       // 12288 B

  gru_prep_kernel<<<1, 192, 0, stream>>>(W_in, b_in, W_ih, W_hh, b_ih, b_hh,
                                         WhhH, WxPackH);

  int n_sim = in_sizes[0] / (NSTEP * INDIM);  // 65536
  int blocks = n_sim / 128;                   // 16 sims/wave * 8 waves/block
  gru_persist_kernel<<<blocks, 256, 0, stream>>>(X, b_hh, W_out, b_out, WhhH,
                                                 WxPackH, out);
}
